// MSATransformer_71614284693906
// MI455X (gfx1250) — matrix operand order repaired and checked by reading, not compiled, not run on hardware
//
#include <hip/hip_runtime.h>

// ---------------- types ----------------
typedef __attribute__((ext_vector_type(16))) __bf16 v16bf;
typedef __attribute__((ext_vector_type(8)))  __bf16 v8bf;
typedef __attribute__((ext_vector_type(8)))  float  v8f;

#define PADTOK 6
#define RR 64
#define CC 256
#define EE 768
#define HH 12
#define DD 64
#define FF 3072
#define NPOS 16384   // RR*CC

// ---------------- helpers ----------------
__device__ __forceinline__ v16bf mk16v(v8bf lo, v8bf hi) {
  v16bf r;
#pragma unroll
  for (int i = 0; i < 8; ++i) { r[i] = lo[i]; r[i + 8] = hi[i]; }
  return r;
}
__device__ __forceinline__ v16bf mk16(const __bf16* p0, const __bf16* p1) {
  return mk16v(*(const v8bf*)p0, *(const v8bf*)p1);
}

// ---------------- embedding gather: src[pos][e] = emb[x[pos]][e] ----------------
__global__ void k_embed(const int* __restrict__ x, const float* __restrict__ emb,
                        float* __restrict__ src) {
  size_t i = (size_t)blockIdx.x * 256 + threadIdx.x;   // NPOS*EE threads
  int e = (int)(i % EE);
  size_t pos = i / EE;
  src[i] = emb[(size_t)x[pos] * EE + e];
}

// ---------------- masks + eff scaling header (single block) ----------------
__global__ void k_mask(const int* __restrict__ x, float* __restrict__ mask0,
                       float* __restrict__ maskFix, float* __restrict__ hdr) {
  __shared__ int colfull[CC];
  __shared__ int cnt0, cnt1;
  int t = threadIdx.x;
  for (int i = t; i < NPOS; i += 256) mask0[i] = (x[i] == PADTOK) ? 1.f : 0.f;
  if (t == 0) { cnt0 = 0; cnt1 = 0; }
  __syncthreads();
  {
    int c = t, s = 0;
    for (int r = 0; r < RR; ++r) s += (x[r * CC + c] == PADTOK);
    colfull[c] = (s == RR);
  }
  __syncthreads();
  for (int i = t; i < NPOS; i += 256) {
    int c = i & (CC - 1);
    maskFix[i] = colfull[c] ? 0.f : mask0[i];
  }
  __syncthreads();
  if (t < RR) {
    int r = t, any0 = 0, any1 = 0;
    for (int c = 0; c < CC; ++c) {
      float m0 = mask0[r * CC + c];
      float mf = colfull[c] ? 0.f : m0;
      if (m0 < 0.5f) any0 = 1;
      if (mf < 0.5f) any1 = 1;
    }
    if (any0) atomicAdd(&cnt0, 1);
    if (any1) atomicAdd(&cnt1, 1);
  }
  __syncthreads();
  if (t == 0) {
    hdr[0] = 0.125f * rsqrtf((float)max(cnt0, 1));  // d^-0.5 * eff^-0.5, layer 0
    hdr[1] = 0.125f * rsqrtf((float)max(cnt1, 1));  // layer 1
  }
}

// ---------------- LayerNorm over E per position, f32 -> bf16 ----------------
__global__ void k_ln(const float* __restrict__ x, const float* __restrict__ g,
                     const float* __restrict__ b, __bf16* __restrict__ y) {
  int pos = blockIdx.x * 8 + (threadIdx.x >> 5);
  int lane = threadIdx.x & 31;
  const float* xr = x + (size_t)pos * EE;
  float s = 0.f, ss = 0.f;
  for (int j = lane; j < EE; j += 32) { float v = xr[j]; s += v; ss += v * v; }
#pragma unroll
  for (int m = 1; m < 32; m <<= 1) { s += __shfl_xor(s, m); ss += __shfl_xor(ss, m); }
  float mean = s * (1.f / EE);
  float var = ss * (1.f / EE) - mean * mean;
  float rstd = rsqrtf(var + 1e-5f);
  __bf16* yr = y + (size_t)pos * EE;
  for (int j = lane; j < EE; j += 32)
    yr[j] = (__bf16)((xr[j] - mean) * rstd * g[j] + b[j]);
}

// ---------------- generic batched WMMA GEMM ----------------
// C[n, m] = sum_k A[m, k] * B[n, k]  (+bias[m]) (+resid) (relu)  out f32 or bf16
// A: [M, K] bf16 row-major.  B: [N, K] bf16 row-major.  ldc = M.
// grid = (N/64, M/128, batch); block = 256 (8 waves, each 16M x 64N tile)
// K loop processes 64 per iteration: all 10 operand load groups are issued
// up front (in-order completion -> partial loadcnt waits), then 8 WMMAs run
// while the younger loads are still in flight. No loop-carried operand regs,
// so loads land directly in the WMMA operand tuples (no register copies).
__global__ __launch_bounds__(256) void k_gemm(
    const __bf16* __restrict__ A, const __bf16* __restrict__ B,
    float* __restrict__ Cf, __bf16* __restrict__ Cb,
    const float* __restrict__ bias, const float* __restrict__ resid,
    int M, int K, int relu,
    long long sA, long long sB, long long sC) {
  int wave = threadIdx.x >> 5, lane = threadIdx.x & 31;
  int half = lane >> 4, lm = lane & 15;
  int mtile = blockIdx.y * 128 + wave * 16;
  int n0 = blockIdx.x * 64;
  A += (size_t)blockIdx.z * sA;
  B += (size_t)blockIdx.z * sB;
  size_t cbase = (size_t)blockIdx.z * sC;

  v8f acc[4] = {};
  const __bf16* arow = A + (size_t)(mtile + lm) * K + (half ? 16 : 0);
  const __bf16* brow = B + (size_t)(n0 + lm) * K + (half ? 16 : 0);
  const size_t bstr = (size_t)16 * K;

  for (int k0 = 0; k0 < K; k0 += 64) {
    // issue all loads for two K-steps
    v16bf a0 = mk16(arow + k0, arow + k0 + 8);
    v16bf a1 = mk16(arow + k0 + 32, arow + k0 + 40);
    v16bf b0[4], b1[4];
#pragma unroll
    for (int t = 0; t < 4; ++t) b0[t] = *(const v16bf*)(brow + t * bstr + k0);
#pragma unroll
    for (int t = 0; t < 4; ++t) b1[t] = *(const v16bf*)(brow + t * bstr + k0 + 32);
    // 8 WMMAs; oldest loads complete first, younger ones overlap with compute
#pragma unroll
    for (int t = 0; t < 4; ++t)
      acc[t] = __builtin_amdgcn_wmma_f32_16x16x32_bf16(
          false, a0, false, b0[t], (short)0, acc[t], false, false);
#pragma unroll
    for (int t = 0; t < 4; ++t)
      acc[t] = __builtin_amdgcn_wmma_f32_16x16x32_bf16(
          false, a1, false, b1[t], (short)0, acc[t], false, false);
  }

  int mbase = mtile + half * 8;
#pragma unroll
  for (int t = 0; t < 4; ++t) {
    int n = n0 + t * 16 + lm;
    size_t co = cbase + (size_t)n * M + mbase;
#pragma unroll
    for (int v = 0; v < 8; ++v) {
      float xv = acc[t][v];
      if (bias) xv += bias[mbase + v];
      if (relu) xv = fmaxf(xv, 0.f);
      if (resid) xv += resid[co + v];
      if (Cf) Cf[co + v] = xv; else Cb[co + v] = (__bf16)xv;
    }
  }
}

// ---------------- row-attn packs ----------------
// dst[h][c][(d*64+r)] = src[(r*C+c)*E + h*64+d] * (opt (1-mask)*scale)
__global__ void k_rowpack_qk(const __bf16* __restrict__ src, __bf16* __restrict__ dst,
                             const float* __restrict__ mask,
                             const float* __restrict__ scaleptr) {
  size_t idx = (size_t)blockIdx.x * 256 + threadIdx.x;  // HH*CC*4096
  int kk = (int)(idx & 4095);
  int c = (int)((idx >> 12) & (CC - 1));
  int h = (int)(idx >> 20);
  int d = kk >> 6, r = kk & 63;
  size_t pos = (size_t)r * CC + c;
  float v = (float)src[pos * EE + h * DD + d];
  if (mask) v *= (1.f - mask[pos]) * (*scaleptr);
  dst[idx] = (__bf16)v;
}

// dst[h][(d*64+r)][ck] = src[(r*C+ck)*E + h*64+d]
__global__ void k_rowpack_v(const __bf16* __restrict__ src, __bf16* __restrict__ dst) {
  size_t idx = (size_t)blockIdx.x * 256 + threadIdx.x;
  int ck = (int)(idx & (CC - 1));
  int kk = (int)((idx >> 8) & 4095);
  int h = (int)(idx >> 20);
  int d = kk >> 6, r = kk & 63;
  dst[idx] = src[((size_t)(r * CC + ck)) * EE + h * DD + d];
}

// softmax over 256 keys; one wave per (h, cq) row; key mask from mask row 0
__global__ void k_softmax_row(const float* __restrict__ S, __bf16* __restrict__ P,
                              const float* __restrict__ maskRow0) {
  int row = blockIdx.x * 8 + (threadIdx.x >> 5);
  int lane = threadIdx.x & 31;
  const float* sr = S + (size_t)row * CC;
  float v[8]; float mx = -1e30f;
#pragma unroll
  for (int j = 0; j < 8; ++j) {
    int ck = j * 32 + lane;
    float xv = sr[ck];
    if (maskRow0[ck] > 0.5f) xv = -1e30f;
    v[j] = xv; mx = fmaxf(mx, xv);
  }
#pragma unroll
  for (int m = 1; m < 32; m <<= 1) mx = fmaxf(mx, __shfl_xor(mx, m));
  float sum = 0.f;
#pragma unroll
  for (int j = 0; j < 8; ++j) { v[j] = expf(v[j] - mx); sum += v[j]; }
#pragma unroll
  for (int m = 1; m < 32; m <<= 1) sum += __shfl_xor(sum, m);
  float inv = 1.f / sum;
  __bf16* pr = P + (size_t)row * CC;
#pragma unroll
  for (int j = 0; j < 8; ++j) pr[j * 32 + lane] = (__bf16)(v[j] * inv);
}

// attn-out repack: AO[(r*C+c)*E + h*64+d] = RO[h][c*4096 + d*64 + r]
__global__ void k_ao_pack(const __bf16* __restrict__ ro, __bf16* __restrict__ ao) {
  size_t idx = (size_t)blockIdx.x * 256 + threadIdx.x;  // NPOS*EE
  int e = (int)(idx % EE);
  size_t pos = idx / EE;
  int h = e / DD, d = e % DD;
  int r = (int)(pos >> 8), c = (int)(pos & (CC - 1));
  ao[idx] = ro[(size_t)h * 1048576 + (size_t)c * 4096 + d * 64 + r];
}

// ---------------- col-attn pack: Qc/Kc [h][c][r][d] (q*0.125), Vc [h][c][d][r] ----------------
__global__ void k_colpack(const __bf16* __restrict__ qb, const __bf16* __restrict__ kb,
                          const __bf16* __restrict__ vb,
                          __bf16* __restrict__ Qc, __bf16* __restrict__ Kc,
                          __bf16* __restrict__ Vc) {
  size_t idx = (size_t)blockIdx.x * 256 + threadIdx.x;  // HH*CC*64*64
  int d = (int)(idx & 63);
  int r = (int)((idx >> 6) & 63);
  int c = (int)((idx >> 12) & (CC - 1));
  int h = (int)(idx >> 20);
  size_t src = ((size_t)(r * CC + c)) * EE + h * DD + d;
  size_t base = ((size_t)(h * CC + c)) * 4096;
  Qc[base + r * 64 + d] = (__bf16)((float)qb[src] * 0.125f);
  Kc[base + r * 64 + d] = kb[src];
  Vc[base + d * 64 + r] = vb[src];
}

// ---------------- fused column attention per (h,c): QK^T -> mask/softmax -> PV ----------------
__global__ __launch_bounds__(128) void k_colattn(
    const __bf16* __restrict__ Qc, const __bf16* __restrict__ Kc,
    const __bf16* __restrict__ Vc, const float* __restrict__ mask,
    __bf16* __restrict__ CO) {
  __shared__ __bf16 pl[4][16 * 64];
  int c = blockIdx.x & (CC - 1);
  int h = blockIdx.x >> 8;
  int wave = threadIdx.x >> 5, lane = threadIdx.x & 31;
  int half = lane >> 4, lm = lane & 15;
  size_t base = ((size_t)(h * CC + c)) * 4096;
  const __bf16* Qh = Qc + base;
  const __bf16* Kh = Kc + base;
  const __bf16* Vh = Vc + base;
  int rq0 = wave * 16;

  // scores: 16 rq x 64 rk
  v8f acc[4] = {};
  {
    const __bf16* arow = Qh + (size_t)(rq0 + lm) * 64 + (half ? 16 : 0);
    const __bf16* brow = Kh + (size_t)lm * 64 + (half ? 16 : 0);
#pragma unroll
    for (int k0 = 0; k0 < 64; k0 += 32) {
      v16bf a = mk16(arow + k0, arow + k0 + 8);
      v16bf bb[4];
#pragma unroll
      for (int t = 0; t < 4; ++t) bb[t] = *(const v16bf*)(brow + (size_t)t * 16 * 64 + k0);
#pragma unroll
      for (int t = 0; t < 4; ++t)
        acc[t] = __builtin_amdgcn_wmma_f32_16x16x32_bf16(
            false, a, false, bb[t], (short)0, acc[t], false, false);
    }
  }
  // key mask over rk
#pragma unroll
  for (int t = 0; t < 4; ++t) {
    float mk = mask[(size_t)(t * 16 + lm) * CC + c];
    if (mk > 0.5f) {
#pragma unroll
      for (int v = 0; v < 8; ++v) acc[t][v] = -1e30f;
    }
  }
  // per-row softmax (row = v + 8*half fixed within 16-lane group; reduce over lanes)
#pragma unroll
  for (int v = 0; v < 8; ++v) {
    float mx = fmaxf(fmaxf(acc[0][v], acc[1][v]), fmaxf(acc[2][v], acc[3][v]));
#pragma unroll
    for (int m = 1; m < 16; m <<= 1) mx = fmaxf(mx, __shfl_xor(mx, m));
    float pv[4]; float sum = 0.f;
#pragma unroll
    for (int t = 0; t < 4; ++t) { pv[t] = expf(acc[t][v] - mx); sum += pv[t]; }
#pragma unroll
    for (int m = 1; m < 16; m <<= 1) sum += __shfl_xor(sum, m);
    float inv = 1.f / sum;
#pragma unroll
    for (int t = 0; t < 4; ++t)
      pl[wave][(v + 8 * half) * 64 + t * 16 + lm] = (__bf16)(pv[t] * inv);
  }
  __syncthreads();
  // PV: out 16 rq x 64 d
  v8f o[4] = {};
  {
    const __bf16* prow = &pl[wave][lm * 64 + (half ? 16 : 0)];
    const __bf16* brow = Vh + (size_t)lm * 64 + (half ? 16 : 0);
#pragma unroll
    for (int k0 = 0; k0 < 64; k0 += 32) {
      v16bf a = mk16(prow + k0, prow + k0 + 8);
      v16bf bb[4];
#pragma unroll
      for (int t = 0; t < 4; ++t) bb[t] = *(const v16bf*)(brow + (size_t)t * 16 * 64 + k0);
#pragma unroll
      for (int t = 0; t < 4; ++t)
        o[t] = __builtin_amdgcn_wmma_f32_16x16x32_bf16(
            false, a, false, bb[t], (short)0, o[t], false, false);
    }
  }
#pragma unroll
  for (int t = 0; t < 4; ++t) {
    int d = t * 16 + lm;
#pragma unroll
    for (int v = 0; v < 8; ++v) {
      int rq = rq0 + 8 * half + v;
      CO[((size_t)(rq * CC + c)) * EE + h * DD + d] = (__bf16)(o[t][v]);
    }
  }
}

// ---------------- misc ----------------
__global__ void k_f2bf(const float* __restrict__ a, __bf16* __restrict__ b) {
  size_t i = (size_t)blockIdx.x * 256 + threadIdx.x;
  b[i] = (__bf16)a[i];
}
// d_out[e*NPOS + pos] = src[pos*E + e]
__global__ void k_outT(const float* __restrict__ src, float* __restrict__ out) {
  size_t i = (size_t)blockIdx.x * 256 + threadIdx.x;  // EE*NPOS
  int e = (int)(i / NPOS);
  int pos = (int)(i % NPOS);
  out[i] = src[(size_t)pos * EE + e];
}

// ---------------- host orchestration ----------------
static inline void launch_gemm(hipStream_t s, const __bf16* A, const __bf16* B,
                               float* Cf, __bf16* Cb, const float* bias,
                               const float* resid, int M, int N, int K, int relu,
                               int batch, long long sA, long long sB, long long sC) {
  dim3 g(N / 64, M / 128, batch);
  k_gemm<<<g, 256, 0, s>>>(A, B, Cf, Cb, bias, resid, M, K, relu, sA, sB, sC);
}

extern "C" void kernel_launch(void* const* d_in, const int* in_sizes, int n_in,
                              void* d_out, int out_size, void* d_ws, size_t ws_size,
                              hipStream_t stream) {
  const int* xtok = (const int*)d_in[0];
  const float* emb = (const float*)d_in[1];
  auto W = [&](int l, int j) { return (const float*)d_in[2 + 26 * l + j]; };

  // ----- workspace carve -----
  char* ws = (char*)d_ws;
  size_t off = 0;
  auto carve = [&](size_t bytes) -> void* {
    void* p = ws + off; off += (bytes + 255) & ~(size_t)255; return p;
  };
  float* hdr     = (float*)carve(256);
  float* mask0   = (float*)carve((size_t)NPOS * 4);
  float* maskFix = (float*)carve((size_t)NPOS * 4);
  float*  SRC = (float*)carve((size_t)NPOS * EE * 4);
  __bf16* LNB = (__bf16*)carve((size_t)NPOS * EE * 2);
  __bf16* QB  = (__bf16*)carve((size_t)NPOS * EE * 2);
  __bf16* KB  = (__bf16*)carve((size_t)NPOS * EE * 2);
  __bf16* VB  = (__bf16*)carve((size_t)NPOS * EE * 2);
  __bf16* PK1 = (__bf16*)carve((size_t)HH * CC * 4096 * 2);
  __bf16* PK2 = (__bf16*)carve((size_t)HH * CC * 4096 * 2);
  __bf16* PK3 = (__bf16*)carve((size_t)HH * CC * 4096 * 2);
  float*  Sbuf = (float*)carve((size_t)HH * CC * CC * 4);
  __bf16* Pbuf = (__bf16*)carve((size_t)HH * CC * CC * 2);
  __bf16* WBF = (__bf16*)carve((size_t)2 * 9437184 * 2);
  // aliases (regions free at time of use; QB..PK1 are contiguous, sizes 256-aligned)
  __bf16* RO  = QB;   // PV output  [h][cq*4096+m], 24 MB
  __bf16* AO  = KB;   // row-attn out in [pos][e]
  __bf16* CO  = KB;   // col-attn out in [pos][e]
  __bf16* FF1 = QB;   // 96 MB spans QB,KB,VB,PK1 exactly

  (void)n_in; (void)in_sizes;
  const int GE = (NPOS * EE) / 256;          // 49152
  const int GP = (HH * CC * 4096) / 256;     // 49152

  // ----- preamble -----
  k_mask<<<1, 256, 0, stream>>>(xtok, mask0, maskFix, hdr);
  k_embed<<<GE, 256, 0, stream>>>(xtok, emb, SRC);

  // weight conversion to bf16
  const int wsz[10] = {EE * EE, EE * EE, EE * EE, EE * EE, EE * EE, EE * EE,
                       EE * EE, EE * EE, FF * EE, EE * FF};
  const int widx[10] = {0, 2, 4, 6, 8, 10, 12, 14, 16, 18};
  __bf16* wp[2][10];
  {
    size_t o2 = 0;
    for (int l = 0; l < 2; ++l)
      for (int j = 0; j < 10; ++j) {
        wp[l][j] = WBF + o2;
        k_f2bf<<<wsz[j] / 256, 256, 0, stream>>>(W(l, widx[j]), wp[l][j]);
        o2 += wsz[j];
      }
  }

  // ----- layers -----
  for (int l = 0; l < 2; ++l) {
    const float* rowMask = (l == 0) ? mask0 : maskFix;
    const float* scalep = hdr + l;

    // --- row attention ---
    k_ln<<<NPOS / 8, 256, 0, stream>>>(SRC, W(l, 20), W(l, 21), LNB);
    launch_gemm(stream, wp[l][0], LNB, nullptr, QB, W(l, 1), nullptr, EE, NPOS, EE, 0, 1, 0, 0, 0);
    launch_gemm(stream, wp[l][1], LNB, nullptr, KB, W(l, 3), nullptr, EE, NPOS, EE, 0, 1, 0, 0, 0);
    launch_gemm(stream, wp[l][2], LNB, nullptr, VB, W(l, 5), nullptr, EE, NPOS, EE, 0, 1, 0, 0, 0);
    k_rowpack_qk<<<GP, 256, 0, stream>>>(QB, PK1, rowMask, scalep);   // Qhat (masked, scaled)
    k_rowpack_qk<<<GP, 256, 0, stream>>>(KB, PK2, nullptr, nullptr);  // Khat
    k_rowpack_v<<<GP, 256, 0, stream>>>(VB, PK3);                     // Vhat
    // S[h][cq][ck] = Khat x Qhat  (tied over rows; K = d*r = 4096)
    launch_gemm(stream, PK2, PK1, Sbuf, nullptr, nullptr, nullptr,
                CC, CC, 4096, 0, HH, 1048576, 1048576, 65536);
    k_softmax_row<<<(HH * CC) / 8, 256, 0, stream>>>(Sbuf, Pbuf, rowMask);
    // RO[h][cq][(d,r)] = Vhat x P
    launch_gemm(stream, PK3, Pbuf, nullptr, RO, nullptr, nullptr,
                4096, CC, CC, 0, HH, 1048576, 65536, 1048576);
    k_ao_pack<<<GE, 256, 0, stream>>>(RO, AO);
    launch_gemm(stream, wp[l][3], AO, SRC, nullptr, W(l, 7), SRC, EE, NPOS, EE, 0, 1, 0, 0, 0);

    // --- column attention (uses fixed-up mask) ---
    k_ln<<<NPOS / 8, 256, 0, stream>>>(SRC, W(l, 22), W(l, 23), LNB);
    launch_gemm(stream, wp[l][4], LNB, nullptr, QB, W(l, 9),  nullptr, EE, NPOS, EE, 0, 1, 0, 0, 0);
    launch_gemm(stream, wp[l][5], LNB, nullptr, KB, W(l, 11), nullptr, EE, NPOS, EE, 0, 1, 0, 0, 0);
    launch_gemm(stream, wp[l][6], LNB, nullptr, VB, W(l, 13), nullptr, EE, NPOS, EE, 0, 1, 0, 0, 0);
    k_colpack<<<GP, 256, 0, stream>>>(QB, KB, VB, PK1, PK2, PK3);
    k_colattn<<<HH * CC, 128, 0, stream>>>(PK1, PK2, PK3, maskFix, CO);
    launch_gemm(stream, wp[l][7], CO, SRC, nullptr, W(l, 15), SRC, EE, NPOS, EE, 0, 1, 0, 0, 0);

    // --- MLP ---
    k_ln<<<NPOS / 8, 256, 0, stream>>>(SRC, W(l, 24), W(l, 25), LNB);
    launch_gemm(stream, wp[l][8], LNB, nullptr, FF1, W(l, 17), nullptr, FF, NPOS, EE, 1, 1, 0, 0, 0);
    launch_gemm(stream, wp[l][9], FF1, SRC, nullptr, W(l, 19), SRC, EE, NPOS, FF, 0, 1, 0, 0, 0);
  }

  // ----- output transpose [pos][e] -> [e][pos] -----
  k_outT<<<GE, 256, 0, stream>>>(SRC, (float*)d_out);
  (void)ws_size; (void)out_size;
}
